// DEQConvergedBaseline_223338299740
// MI455X (gfx1250) — matrix operand order repaired and checked by reading, not compiled, not run on hardware
//
#include <hip/hip_runtime.h>
#include <hip/hip_bf16.h>

// ---------------------------------------------------------------------------
// Types for CDNA5 WMMA bf16 path
// ---------------------------------------------------------------------------
typedef __attribute__((ext_vector_type(16))) __bf16 bf16x16;
typedef __attribute__((ext_vector_type(8)))  __bf16 bf16x8;
typedef __attribute__((ext_vector_type(8)))  float  f32x8;
typedef __attribute__((ext_vector_type(4)))  unsigned int u32x4;
typedef __attribute__((ext_vector_type(8)))  int   i32x8;
typedef __attribute__((ext_vector_type(4)))  int   i32x4;

union FragAB {
    bf16x16 v;
    bf16x8  h[2];
};

#if defined(__has_builtin)
#  if __has_builtin(__builtin_amdgcn_tensor_load_to_lds)
#    define HAS_TDM 1
#  endif
#endif

// LDS tile geometry: 128 rows x 64 bf16 (128 B) + 16 B pad -> 144 B pitch
#define TILE_ROWS   128
#define TILE_K      64
#define LDS_PITCH   144
#define LDS_TILE_SZ (TILE_ROWS * LDS_PITCH)

__device__ __forceinline__ unsigned lds_off(const void* p)
{
    // generic shared pointers carry the LDS byte offset in the low 32 bits
    return (unsigned)(unsigned long long)(uintptr_t)p;
}

#ifdef HAS_TDM
// ---------------------------------------------------------------------------
// TDM: async DMA of a 2-D bf16 tile (TILE_ROWS x TILE_K, row stride = ld elts)
// from global to LDS, inserting 16 B padding after every 128 B row.
// D# per CDNA5 ISA ch.8 (group0: count/lds/global/type; group1: dims/strides).
// ---------------------------------------------------------------------------
__device__ __forceinline__ void tdm_stage_tile(const __bf16* gsrc, unsigned lds_addr, int ld)
{
    unsigned long long ga = (unsigned long long)(uintptr_t)gsrc;
    u32x4 g0;
    g0[0] = 1u;                                                // count=1, no gather
    g0[1] = lds_addr;                                          // lds_addr
    g0[2] = (unsigned)ga;                                      // global_addr[31:0]
    g0[3] = (unsigned)((ga >> 32) & 0x01FFFFFFu) | (2u << 30); // global_addr[56:32] | type=2
    i32x8 g1;
    // wg_mask=0 | data_size=1(2B)<<16 | pad_enable<<20 | pad_interval=4(32 dw)<<22
    // | pad_amount=3(4 dw = 16B)<<25
    g1[0] = (int)((1u << 16) | (1u << 20) | (4u << 22) | (3u << 25));
    g1[1] = (int)(((unsigned)TILE_K) << 16);        // barrier_addr=0 | tensor_dim0[15:0]
    g1[2] = (int)(((unsigned)TILE_ROWS) << 16);     // tensor_dim0[31:16]=0 | tensor_dim1[15:0]
    g1[3] = (int)(((unsigned)TILE_K) << 16);        // tensor_dim1[31:16]=0 | tile_dim0
    g1[4] = (int)((unsigned)TILE_ROWS);             // tile_dim1 | tile_dim2=0
    g1[5] = ld;                                     // tensor_dim0_stride[31:0] (elts)
    g1[6] = 0;                                      // stride0[47:32] | stride1 lo
    g1[7] = 0;                                      // stride1 hi
    i32x4 g2 = {0, 0, 0, 0};
    i32x4 g3 = {0, 0, 0, 0};
#if __clang_major__ >= 23
    i32x8 g4 = {0, 0, 0, 0, 0, 0, 0, 0};
    __builtin_amdgcn_tensor_load_to_lds(g0, g1, g2, g3, g4, 0);
#else
    __builtin_amdgcn_tensor_load_to_lds(g0, g1, g2, g3, 0);
#endif
}
#else
// ---------------------------------------------------------------------------
// Fallback cooperative staging (all 256 threads): global b128 -> LDS b128
// ---------------------------------------------------------------------------
__device__ __forceinline__ void coop_stage_tile(const __bf16* gsrc, unsigned char* dst, int ld)
{
    for (int t = threadIdx.x; t < TILE_ROWS * 4; t += 256) {
        const int row = t >> 2, seg = t & 3;
        *(bf16x8*)(dst + row * LDS_PITCH + seg * 16) =
            *(const bf16x8*)((const unsigned char*)(gsrc + (size_t)row * ld) + seg * 16);
    }
}
#endif

// ---------------------------------------------------------------------------
// fp32 -> bf16 elementwise convert (grid-stride)
// ---------------------------------------------------------------------------
__global__ __launch_bounds__(256)
void f32_to_bf16_kernel(const float* __restrict__ in, __bf16* __restrict__ out, int n)
{
    int i = blockIdx.x * 256 + threadIdx.x;
    int stride = gridDim.x * 256;
    for (; i < n; i += stride) {
        out[i] = (__bf16)in[i];
    }
}

// ---------------------------------------------------------------------------
// NT GEMM:  C[M,N] = A[M,K] * B[N,K]^T   (bf16 operands, fp32 accumulate)
// Block tile 128x128, 8 waves (32x64 each = 2x4 WMMA tiles), K-step 64.
// Operands staged to LDS by the Tensor Data Mover, double buffered against
// the WMMA loop (TENSORcnt + workgroup barrier).  M,N mult of 128; K of 64.
// ---------------------------------------------------------------------------
__global__ __launch_bounds__(256)
void gemm_nt_bf16_kernel(const __bf16* __restrict__ A, const __bf16* __restrict__ B,
                         __bf16* __restrict__ Cb, float* __restrict__ Cf,
                         int K, int lda, int ldb, int ldc)
{
    __shared__ __align__(128) unsigned char ldsA[2][LDS_TILE_SZ];
    __shared__ __align__(128) unsigned char ldsB[2][LDS_TILE_SZ];

    const int lane = threadIdx.x & 31;
    const int wid  = threadIdx.x >> 5;      // 0..7
    const int r    = lane & 15;             // row/col within 16-wide tile
    const int half = lane >> 4;             // 0 or 1 (lane group)

    const int mw = (wid >> 1) * 32;         // wave M origin within block tile
    const int nw = (wid & 1) * 64;          // wave N origin within block tile

    const __bf16* Ablk = A + (size_t)blockIdx.y * 128 * lda;
    const __bf16* Bblk = B + (size_t)blockIdx.x * 128 * ldb;

    f32x8 acc[2][4];
    const f32x8 zero = {0.f, 0.f, 0.f, 0.f, 0.f, 0.f, 0.f, 0.f};
#pragma unroll
    for (int i = 0; i < 2; ++i)
#pragma unroll
        for (int j = 0; j < 4; ++j)
            acc[i][j] = zero;

    // ---- prologue: stage k0 = 0 slabs into buffer 0 ----
#ifdef HAS_TDM
    if (wid == 0) {
        tdm_stage_tile(Ablk, lds_off(&ldsA[0][0]), lda);
        tdm_stage_tile(Bblk, lds_off(&ldsB[0][0]), ldb);
        __builtin_amdgcn_s_wait_tensorcnt(0);
    }
#else
    coop_stage_tile(Ablk, &ldsA[0][0], lda);
    coop_stage_tile(Bblk, &ldsB[0][0], ldb);
#endif
    __syncthreads();

    int buf = 0;
    for (int k0 = 0; k0 < K; k0 += TILE_K) {
        const int nbuf = buf ^ 1;
        // kick off next slab DMA (overlaps with this k-step's WMMA work)
        if (k0 + TILE_K < K) {
#ifdef HAS_TDM
            if (wid == 0) {
                tdm_stage_tile(Ablk + k0 + TILE_K, lds_off(&ldsA[nbuf][0]), lda);
                tdm_stage_tile(Bblk + k0 + TILE_K, lds_off(&ldsB[nbuf][0]), ldb);
            }
#else
            coop_stage_tile(Ablk + k0 + TILE_K, &ldsA[nbuf][0], lda);
            coop_stage_tile(Bblk + k0 + TILE_K, &ldsB[nbuf][0], ldb);
#endif
        }

        const unsigned char* baA = &ldsA[buf][0];
        const unsigned char* baB = &ldsB[buf][0];
#pragma unroll
        for (int kk = 0; kk < TILE_K; kk += 32) {
            // A fragments (16x32 bf16, ISA 7.12.2 lane layout):
            //   lanes 0-15 : row r, K = kk+0..7 and kk+16..23
            //   lanes 16-31: row r, K = kk+8..15 and kk+24..31
            FragAB a[2];
#pragma unroll
            for (int i = 0; i < 2; ++i) {
                const unsigned char* pa =
                    baA + (mw + i * 16 + r) * LDS_PITCH + (kk + half * 8) * 2;
                a[i].h[0] = *(const bf16x8*)(pa);
                a[i].h[1] = *(const bf16x8*)(pa + 32);
            }
            // B fragments (32x16 of B^T, same K layout as A):
            //   lanes 0-15 : col r, K = kk+0..7 and kk+16..23 ; lanes 16-31: K = kk+8..15 and kk+24..31
            FragAB b[4];
#pragma unroll
            for (int j = 0; j < 4; ++j) {
                const unsigned char* pb =
                    baB + (nw + j * 16 + r) * LDS_PITCH + (kk + half * 8) * 2;
                b[j].h[0] = *(const bf16x8*)(pb);
                b[j].h[1] = *(const bf16x8*)(pb + 32);
            }
#pragma unroll
            for (int i = 0; i < 2; ++i)
#pragma unroll
                for (int j = 0; j < 4; ++j)
                    acc[i][j] = __builtin_amdgcn_wmma_f32_16x16x32_bf16(
                        false, a[i].v, false, b[j].v,
                        (short)0, acc[i][j], false, false);
        }

        // publish next buffer: DMA must land, all waves must be done reading
#ifdef HAS_TDM
        if (wid == 0) __builtin_amdgcn_s_wait_tensorcnt(0);
#endif
        __syncthreads();
        buf = nbuf;
    }

    // C/D 16x16 f32 layout: VGPR e -> M = e + half*8, N = r (within tile)
    const int m0 = blockIdx.y * 128 + mw;
    const int n0 = blockIdx.x * 128 + nw;
#pragma unroll
    for (int i = 0; i < 2; ++i) {
#pragma unroll
        for (int j = 0; j < 4; ++j) {
#pragma unroll
            for (int e = 0; e < 8; ++e) {
                const int row = m0 + i * 16 + half * 8 + e;
                const int col = n0 + j * 16 + r;
                const float val = acc[i][j][e];
                if (Cb) Cb[(size_t)row * ldc + col] = (__bf16)val;
                if (Cf) Cf[(size_t)row * ldc + col] = val;
            }
        }
    }
}

// ---------------------------------------------------------------------------
// Row softmax of beta * S, S fp32 [rows, ncols], output bf16 W.
// One block (256 threads) per row; ncols == 4096 -> 16 elements/thread.
// ---------------------------------------------------------------------------
__global__ __launch_bounds__(256)
void softmax_rows_kernel(const float* __restrict__ S, __bf16* __restrict__ W,
                         const float* __restrict__ log_beta, int ncols)
{
    __shared__ float red[256];
    const int row = blockIdx.x;
    const int tid = threadIdx.x;
    const float* srow = S + (size_t)row * ncols;
    __bf16* wrow = W + (size_t)row * ncols;
    const float beta = __expf(log_beta[0]);

    // pass 1: max
    float lmax = -3.402823466e38f;
    for (int c = tid; c < ncols; c += 256) lmax = fmaxf(lmax, srow[c]);
    red[tid] = lmax;
    __syncthreads();
    for (int s = 128; s > 0; s >>= 1) {
        if (tid < s) red[tid] = fmaxf(red[tid], red[tid + s]);
        __syncthreads();
    }
    const float rmax = red[0];
    __syncthreads();

    // pass 2: exp + sum (exponentials kept in registers: 4096/256 = 16)
    float ev[16];
    float lsum = 0.f;
    int t = 0;
    for (int c = tid; c < ncols; c += 256, ++t) {
        const float e = __expf(beta * (srow[c] - rmax));
        ev[t] = e;
        lsum += e;
    }
    red[tid] = lsum;
    __syncthreads();
    for (int s = 128; s > 0; s >>= 1) {
        if (tid < s) red[tid] += red[tid + s];
        __syncthreads();
    }
    const float inv = 1.0f / red[0];

    // pass 3: normalize, store bf16
    t = 0;
    for (int c = tid; c < ncols; c += 256, ++t) {
        wrow[c] = (__bf16)(ev[t] * inv);
    }
}

// ---------------------------------------------------------------------------
// Host-side orchestration
// ---------------------------------------------------------------------------
extern "C" void kernel_launch(void* const* d_in, const int* in_sizes, int n_in,
                              void* d_out, int out_size, void* d_ws, size_t ws_size,
                              hipStream_t stream)
{
    (void)in_sizes; (void)n_in; (void)out_size; (void)ws_size;

    const float* query    = (const float*)d_in[0];  // [B, D]
    const float* patterns = (const float*)d_in[1];  // [N, D]
    const float* W_q      = (const float*)d_in[2];  // [D, D]
    const float* W_k      = (const float*)d_in[3];  // [D, D]
    const float* W_v      = (const float*)d_in[4];  // [D, D]
    const float* log_beta = (const float*)d_in[5];  // scalar

    const int Brows = 2048, Nrows = 4096, D = 1024;
    const int NITER = 31;   // z0 construction + 30 scan steps

    // workspace carve-out (all buffers fully written before read)
    char* ws = (char*)d_ws;
    size_t off = 0;
    auto take = [&](size_t bytes) -> char* {
        char* p = ws + off;
        off += (bytes + 255) & ~(size_t)255;
        return p;
    };
    __bf16* qry16 = (__bf16*)take((size_t)Brows * D * 2);
    __bf16* pat16 = (__bf16*)take((size_t)Nrows * D * 2);
    __bf16* wq16  = (__bf16*)take((size_t)D * D * 2);
    __bf16* wk16  = (__bf16*)take((size_t)D * D * 2);
    __bf16* wv16  = (__bf16*)take((size_t)D * D * 2);
    __bf16* kmat  = (__bf16*)take((size_t)Nrows * D * 2);   // k,   [N, D] row-major
    __bf16* vt    = (__bf16*)take((size_t)D * Nrows * 2);   // v^T, [D, N] row-major
    __bf16* zbuf  = (__bf16*)take((size_t)Brows * D * 2);
    __bf16* qpbuf = (__bf16*)take((size_t)Brows * D * 2);
    float*  sbuf  = (float*) take((size_t)Brows * Nrows * 4);
    __bf16* wbuf  = (__bf16*)take((size_t)Brows * Nrows * 2);

    const dim3 blk(256);

    // --- convert fp32 inputs to bf16 ---
    f32_to_bf16_kernel<<<dim3(2048), blk, 0, stream>>>(query,    qry16, Brows * D);
    f32_to_bf16_kernel<<<dim3(2048), blk, 0, stream>>>(patterns, pat16, Nrows * D);
    f32_to_bf16_kernel<<<dim3(1024), blk, 0, stream>>>(W_q,      wq16,  D * D);
    f32_to_bf16_kernel<<<dim3(1024), blk, 0, stream>>>(W_k,      wk16,  D * D);
    f32_to_bf16_kernel<<<dim3(1024), blk, 0, stream>>>(W_v,      wv16,  D * D);

    // --- k = patterns @ W_k^T  : NT, M=N(4096), N=D, K=D -> bf16 [N, D] ---
    gemm_nt_bf16_kernel<<<dim3(D / 128, Nrows / 128), blk, 0, stream>>>(
        pat16, wk16, kmat, nullptr, D, D, D, D);

    // --- v^T = W_v @ patterns^T : NT, M=D, N=N(4096), K=D -> bf16 [D, N] ---
    gemm_nt_bf16_kernel<<<dim3(Nrows / 128, D / 128), blk, 0, stream>>>(
        wv16, pat16, vt, nullptr, D, D, D, Nrows);

    // --- fixed-point iterations ---
    for (int it = 0; it < NITER; ++it) {
        const __bf16* zin = (it == 0) ? qry16 : zbuf;

        // qp = z @ W_q^T : NT, M=B, N=D, K=D -> bf16
        gemm_nt_bf16_kernel<<<dim3(D / 128, Brows / 128), blk, 0, stream>>>(
            zin, wq16, qpbuf, nullptr, D, D, D, D);

        // s = qp @ k^T : NT, M=B, N=N(4096), K=D -> fp32
        gemm_nt_bf16_kernel<<<dim3(Nrows / 128, Brows / 128), blk, 0, stream>>>(
            qpbuf, kmat, nullptr, sbuf, D, D, D, Nrows);

        // w = softmax(beta * s) -> bf16
        softmax_rows_kernel<<<dim3(Brows), blk, 0, stream>>>(sbuf, wbuf, log_beta, Nrows);

        // z = w @ v = w @ (v^T)^T : NT, M=B, N=D, K=N(4096) -> bf16 (+fp32 out on last)
        float* outf = (it == NITER - 1) ? (float*)d_out : nullptr;
        gemm_nt_bf16_kernel<<<dim3(D / 128, Brows / 128), blk, 0, stream>>>(
            wbuf, vt, zbuf, outf, Nrows, Nrows, Nrows, D);
    }
}
